// SubAgentSystem_46608985096880
// MI455X (gfx1250) — matrix operand order repaired and checked by reading, not compiled, not run on hardware
//
#include <hip/hip_runtime.h>
#include <hip/hip_bf16.h>
#include <math.h>

// Problem constants (match reference)
#define BB   128
#define SS   2048
#define DD   512
#define AA   3
#define SBLK 32          // seq rows per block
#define DW   (DD/2)      // row length in u32 (packed bf16 pairs) = 256

typedef __attribute__((ext_vector_type(16))) __bf16 v16bf;
typedef __attribute__((ext_vector_type(8)))  float  v8f;

union FragU { unsigned u[8]; v16bf bf; };

// ---------- bf16 split helpers (round-to-nearest-even) ----------
__device__ __forceinline__ unsigned bf16_rne(float f) {
    unsigned u = __float_as_uint(f);
    unsigned r = u + 0x7FFFu + ((u >> 16) & 1u);
    return r >> 16;
}
__device__ __forceinline__ float bf16_to_f(unsigned b) {
    return __uint_as_float(b << 16);
}
__device__ __forceinline__ void split2(float f, unsigned& hi, unsigned& lo) {
    hi = bf16_rne(f);
    lo = bf16_rne(f - bf16_to_f(hi));
}

__device__ __forceinline__ v8f wmma_bf16(const FragU& a, const FragU& b, v8f c) {
    return __builtin_amdgcn_wmma_f32_16x16x32_bf16(
        /*neg_a=*/false, a.bf, /*neg_b=*/false, b.bf,
        /*c_mod=*/(short)0, c, /*reuse_a=*/false, /*reuse_b=*/false);
}

// ---------------------------------------------------------------
// Kernel 1: mean-pool over S, 4-way routing, argmax -> aid[b]
// ---------------------------------------------------------------
__global__ __launch_bounds__(256) void route_kernel(
    const float* __restrict__ h, const float* __restrict__ sel_w,
    const float* __restrict__ sel_b, int* __restrict__ aid)
{
    __shared__ float pooled[DD];
    __shared__ float logits[AA + 1];
    const int b = blockIdx.x;
    const int t = threadIdx.x;

    // Each thread owns 2 consecutive feature columns (float2 coalesced)
    const float* hb = h + (size_t)b * SS * DD + 2 * t;
    float ax = 0.f, ay = 0.f;
    for (int s = 0; s < SS; ++s) {
        float2 v = *(const float2*)(hb + (size_t)s * DD);
        ax += v.x; ay += v.y;
    }
    const float inv = 1.0f / (float)SS;
    pooled[2 * t]     = ax * inv;
    pooled[2 * t + 1] = ay * inv;
    __syncthreads();

    if (t < 32) {  // wave 0 computes all A+1 logits (wave32 reductions)
        #pragma unroll
        for (int a = 0; a < AA + 1; ++a) {
            float sum = 0.f;
            for (int d = t; d < DD; d += 32) sum += pooled[d] * sel_w[a * DD + d];
            #pragma unroll
            for (int o = 16; o > 0; o >>= 1) sum += __shfl_xor(sum, o, 32);
            if (t == 0) logits[a] = sum + sel_b[a];
        }
    }
    __syncthreads();
    if (t == 0) {
        int best = 0; float bv = logits[0];
        #pragma unroll
        for (int a = 1; a < AA + 1; ++a)
            if (logits[a] > bv) { bv = logits[a]; best = a; }
        aid[b] = best;
    }
}

// ---------------------------------------------------------------
// Kernel 2: pre-split f32 weights into bf16 hi/lo planes (w1 & w2)
// ---------------------------------------------------------------
__global__ __launch_bounds__(256) void split_kernel(
    const float* __restrict__ w1, const float* __restrict__ w2,
    unsigned short* __restrict__ w1h, unsigned short* __restrict__ w1l,
    unsigned short* __restrict__ w2h, unsigned short* __restrict__ w2l)
{
    const int n = AA * DD * DD;
    int i = blockIdx.x * 256 + threadIdx.x;
    unsigned hi, lo;
    if (i < n) {
        split2(w1[i], hi, lo);
        w1h[i] = (unsigned short)hi; w1l[i] = (unsigned short)lo;
    } else {
        int j = i - n;
        split2(w2[j], hi, lo);
        w2h[j] = (unsigned short)hi; w2l[j] = (unsigned short)lo;
    }
}

// ---------------------------------------------------------------
// bf16x3 GEMM tile: C[16x128] += A[16x512] * B[512x128]
//   A: LDS u32 planes (packed bf16 pairs), row stride DW
//   B: global u32 planes of pre-split weights; B[k][n] = W[n][k],
//      so lane n reads its W row contiguously.
// A-frag layout (16-bit A 16x32, ISA 7.12.2): lane m = lane&15,
//   VGPR r holds K pair at kb = 2r + 8*(r>>2) + 8*(lane>=16).
// B-frag layout (32x16): same K order as A-frag,
//   VGPR r holds K pair at kb = 2r + 8*(r>>2) + 8*(lane>=16).
// ---------------------------------------------------------------
__device__ __forceinline__ void gemm_tile(
    const unsigned* __restrict__ Ah, const unsigned* __restrict__ Al,
    const unsigned* __restrict__ Bh, const unsigned* __restrict__ Bl,
    int m0, int n0, int lane, v8f* acc)
{
    const int lm = lane & 15;
    const int half = lane >> 4;
    const unsigned* arow_h = Ah + (m0 + lm) * DW;
    const unsigned* arow_l = Al + (m0 + lm) * DW;

    for (int kk = 0; kk < DD / 32; ++kk) {
        FragU ahf, alf;
        #pragma unroll
        for (int r = 0; r < 8; ++r) {
            int off = kk * 16 + r + 4 * (r >> 2) + 4 * half;
            ahf.u[r] = arow_h[off];
            alf.u[r] = arow_l[off];
        }
        #pragma unroll
        for (int nt = 0; nt < 8; ++nt) {
            const unsigned* brow_h = Bh + (size_t)(n0 + nt * 16 + lm) * DW + kk * 16 + 4 * half;
            const unsigned* brow_l = Bl + (size_t)(n0 + nt * 16 + lm) * DW + kk * 16 + 4 * half;
            FragU bhf, blf;
            #pragma unroll
            for (int r = 0; r < 8; ++r) { bhf.u[r] = brow_h[r + 4 * (r >> 2)]; blf.u[r] = brow_l[r + 4 * (r >> 2)]; }
            acc[nt] = wmma_bf16(ahf, bhf, acc[nt]);  // hi*hi
            acc[nt] = wmma_bf16(ahf, blf, acc[nt]);  // hi*lo
            acc[nt] = wmma_bf16(alf, bhf, acc[nt]);  // lo*hi
        }
    }
}

// ---------------------------------------------------------------
// Kernel 3: per (batch, 32-row seq tile) fused MLP
//   out = h + gelu(h*W1^T)*W2^T  (or out = h if aid==0)
// 8 waves: wave w -> rows 16*(w>>2), cols 128*(w&3)
// LDS: h_hi | h_lo | hid_hi | hid_lo, each 32x512 bf16 = 32KB (128KB total)
// ---------------------------------------------------------------
__global__ __launch_bounds__(256) void agent_kernel(
    const float* __restrict__ h, float* __restrict__ out,
    const int* __restrict__ aid,
    const unsigned* __restrict__ w1h, const unsigned* __restrict__ w1l,
    const unsigned* __restrict__ w2h, const unsigned* __restrict__ w2l)
{
    const int b = blockIdx.y;
    const int tile = blockIdx.x;
    const int t = threadIdx.x;

    const float* __restrict__ hsrc = h   + ((size_t)b * SS + tile * SBLK) * DD;
    float* __restrict__       osrc = out + ((size_t)b * SS + tile * SBLK) * DD;

    const int a = aid[b];
    if (a == 0) {  // uniform per block: plain streaming copy
        const float4* h4 = (const float4*)hsrc;
        float4* o4 = (float4*)osrc;
        #pragma unroll
        for (int j = 0; j < 16; ++j) o4[t + 256 * j] = h4[t + 256 * j];
        return;
    }
    const size_t wofs = (size_t)(a - 1) * (DD * DD / 2);  // u32 elements
    const unsigned* B1h = w1h + wofs; const unsigned* B1l = w1l + wofs;
    const unsigned* B2h = w2h + wofs; const unsigned* B2l = w2l + wofs;

    extern __shared__ unsigned char smem[];
    unsigned* hh = (unsigned*)smem;          // h hi   [32][256] u32
    unsigned* hl = hh + SBLK * DW;           // h lo
    unsigned* gh = hl + SBLK * DW;           // hid hi
    unsigned* gl = gh + SBLK * DW;           // hid lo

    // ---- stage h tile: f32 -> (hi,lo) bf16 planes, packed pairs ----
    #pragma unroll
    for (int j = 0; j < 16; ++j) {
        int i4 = t + 256 * j;                 // float4 index in [0,4096)
        float4 v = ((const float4*)hsrc)[i4];
        unsigned h0, l0, h1, l1, h2, l2, h3, l3;
        split2(v.x, h0, l0); split2(v.y, h1, l1);
        split2(v.z, h2, l2); split2(v.w, h3, l3);
        int base = i4 * 4;
        int row = base >> 9;                  // /512
        int cw  = (base & 511) >> 1;          // u32 column
        hh[row * DW + cw]     = h0 | (h1 << 16);
        hh[row * DW + cw + 1] = h2 | (h3 << 16);
        hl[row * DW + cw]     = l0 | (l1 << 16);
        hl[row * DW + cw + 1] = l2 | (l3 << 16);
    }
    __syncthreads();

    const int wv = t >> 5, lane = t & 31;
    const int m0 = (wv >> 2) * 16;
    const int n0 = (wv & 3) * 128;
    const int lm = lane & 15, half = lane >> 4;

    // ---- stage 1: hid = gelu(h * W1^T) ----
    v8f acc[8];
    #pragma unroll
    for (int nt = 0; nt < 8; ++nt)
        #pragma unroll
        for (int v = 0; v < 8; ++v) acc[nt][v] = 0.0f;

    gemm_tile(hh, hl, B1h, B1l, m0, n0, lane, acc);

    unsigned short* gh16 = (unsigned short*)gh;
    unsigned short* gl16 = (unsigned short*)gl;
    #pragma unroll
    for (int nt = 0; nt < 8; ++nt) {
        #pragma unroll
        for (int v = 0; v < 8; ++v) {
            float x = acc[nt][v];
            float g = 0.5f * x * (1.0f + erff(x * 0.70710678118654752f));
            unsigned ghi, glo; split2(g, ghi, glo);
            int m = m0 + v + 8 * half;        // C layout: VGPR v -> M = v (+8 hi half)
            int col = n0 + nt * 16 + lm;      // N = lane&15
            gh16[m * DD + col] = (unsigned short)ghi;
            gl16[m * DD + col] = (unsigned short)glo;
        }
    }
    __syncthreads();

    // ---- stage 2: delta = hid * W2^T ; out = h + delta ----
    v8f acc2[8];
    #pragma unroll
    for (int nt = 0; nt < 8; ++nt)
        #pragma unroll
        for (int v = 0; v < 8; ++v) acc2[nt][v] = 0.0f;

    gemm_tile(gh, gl, B2h, B2l, m0, n0, lane, acc2);

    const unsigned short* hh16 = (const unsigned short*)hh;
    const unsigned short* hl16 = (const unsigned short*)hl;
    #pragma unroll
    for (int nt = 0; nt < 8; ++nt) {
        #pragma unroll
        for (int v = 0; v < 8; ++v) {
            int m = m0 + v + 8 * half;
            int col = n0 + nt * 16 + lm;
            float base = bf16_to_f((unsigned)hh16[m * DD + col]) +
                         bf16_to_f((unsigned)hl16[m * DD + col]);
            osrc[(size_t)m * DD + col] = base + acc2[nt][v];
        }
    }
}

// ---------------------------------------------------------------
extern "C" void kernel_launch(void* const* d_in, const int* in_sizes, int n_in,
                              void* d_out, int out_size, void* d_ws, size_t ws_size,
                              hipStream_t stream)
{
    const float* h     = (const float*)d_in[0];
    const float* sel_w = (const float*)d_in[1];
    const float* sel_b = (const float*)d_in[2];
    const float* w1    = (const float*)d_in[3];
    const float* w2    = (const float*)d_in[4];
    float* out = (float*)d_out;

    // Workspace layout: [aid int[128]] pad.. [w1h][w1l][w2h][w2l] bf16 planes
    char* ws = (char*)d_ws;
    int* aid = (int*)ws;
    const size_t NW = (size_t)AA * DD * DD;               // 786432
    unsigned short* w1h = (unsigned short*)(ws + 4096);
    unsigned short* w1l = w1h + NW;
    unsigned short* w2h = w1l + NW;
    unsigned short* w2l = w2h + NW;

    route_kernel<<<BB, 256, 0, stream>>>(h, sel_w, sel_b, aid);

    int nsplit = (int)((2 * NW) / 256);
    split_kernel<<<nsplit, 256, 0, stream>>>(w1, w2, w1h, w1l, w2h, w2l);

    dim3 grid(SS / SBLK, BB);   // 64 x 128
    size_t lds = (size_t)4 * SBLK * DW * sizeof(unsigned);  // 128 KB
    agent_kernel<<<grid, 256, lds, stream>>>(
        h, out, aid,
        (const unsigned*)w1h, (const unsigned*)w1l,
        (const unsigned*)w2h, (const unsigned*)w2l);
}
